// Drugemb_3350074491412
// MI455X (gfx1250) — matrix operand order repaired and checked by reading, not compiled, not run on hardware
//
#include <hip/hip_runtime.h>
#include <math.h>

// ---------------------------------------------------------------------------
// CDNA5 (gfx1250) wave32 WMMA types
// ---------------------------------------------------------------------------
typedef __bf16 bf16;
typedef bf16  v16bf __attribute__((ext_vector_type(16)));
typedef float v8f   __attribute__((ext_vector_type(8)));

// A-matrix 16x32 bf16 lane layout (CDNA5 ISA 7.12.2):
//   lane = half*16 + m;  element j -> k = half*16 + j (contiguous order, same as B; not the ISA table order)
//   => two contiguous 8-float runs per lane per k-step (2x 32B global loads).
// B-matrix 32x16 bf16: col = lane&15; element j -> k = half*16 + j
//   => with W transposed in LDS, one contiguous 32B LDS load (2x ds_load_b128).
// C/D v8f: row = tile_m + half*8 + r, col = lane&15.

__device__ __forceinline__ void load_afrag(v16bf* af, const float* __restrict__ ar,
                                           int half, float rs, int KT) {
  // caller fully unrolls via constexpr KT at each call site
}

// ---------------------------------------------------------------------------
// Dual fused GEMM (one GCN layer):
//   out = relu((A1 .* rowscale) @ Wg + bg) + relu(A2 @ Wr + br)
// A1/A2 fp32 [M,KP] row-major zero-padded cols Kreal..KP-1 ; KP multiple of 32.
// Both weights staged transposed+padded to LDS as bf16.
// Block = 256 threads = 8 waves; wave owns a 16-row tile, sweeps N/16 tiles.
// ---------------------------------------------------------------------------
template <int KP, int N>
__global__ __launch_bounds__(256) void wmma_dual_gemm(
    const float* __restrict__ A1, const float* __restrict__ rowscale,
    const float* __restrict__ Wg, const float* __restrict__ bg,
    const float* __restrict__ A2,
    const float* __restrict__ Wr, const float* __restrict__ br,
    float* __restrict__ out, int M, int Kreal)
{
  constexpr int KT = KP / 32;
  constexpr int NT = N / 16;
  __shared__ __align__(32) bf16 sW[2 * N * KP];   // [0]: Wg^T, [1]: Wr^T

  for (int i = threadIdx.x; i < N * KP; i += 256) {
    int n = i / KP, k = i % KP;
    bool kv = k < Kreal;
    sW[i]          = kv ? (bf16)Wg[(size_t)k * N + n] : (bf16)0.0f;
    sW[N * KP + i] = kv ? (bf16)Wr[(size_t)k * N + n] : (bf16)0.0f;
  }
  __syncthreads();

  const int wave   = threadIdx.x >> 5;
  const int lane   = threadIdx.x & 31;
  const int half   = lane >> 4;
  const int lrow   = lane & 15;
  const int tile_m = blockIdx.x * 128 + wave * 16;
  if (tile_m >= M) return;   // wave-uniform: EXEC all-ones around WMMA

  const int  row = tile_m + lrow;
  const bool rv  = row < M;
  const float rs = rv ? rowscale[row] : 1.0f;
  const float* ar1 = A1 + (size_t)(rv ? row : 0) * KP;
  const float* ar2 = A2 + (size_t)(rv ? row : 0) * KP;

  v16bf af1[KT], af2[KT];
#pragma unroll
  for (int kt = 0; kt < KT; ++kt) {
    const int b0 = kt * 32 + half * 16;
    v8f xa = *(const v8f*)(ar1 + b0);
    v8f xb = *(const v8f*)(ar1 + b0 + 8);
    v8f ya = *(const v8f*)(ar2 + b0);
    v8f yb = *(const v8f*)(ar2 + b0 + 8);
#pragma unroll
    for (int j = 0; j < 8; ++j) {
      af1[kt][j]     = (bf16)(xa[j] * rs);
      af1[kt][j + 8] = (bf16)(xb[j] * rs);
      af2[kt][j]     = (bf16)ya[j];
      af2[kt][j + 8] = (bf16)yb[j];
    }
  }

  const bool full = (tile_m + 16) <= M;   // wave-uniform

#pragma unroll 1
  for (int nt = 0; nt < NT; ++nt) {
    const int col = nt * 16 + lrow;
    const bf16* bp1 = sW + (size_t)col * KP;
    const bf16* bp2 = bp1 + N * KP;
    v8f acc1 = {0.f, 0.f, 0.f, 0.f, 0.f, 0.f, 0.f, 0.f};
    v8f acc2 = {0.f, 0.f, 0.f, 0.f, 0.f, 0.f, 0.f, 0.f};
#pragma unroll
    for (int kt = 0; kt < KT; ++kt) {
      v16bf bf1 = *(const v16bf*)(bp1 + kt * 32 + half * 16);
      v16bf bf2 = *(const v16bf*)(bp2 + kt * 32 + half * 16);
      acc1 = __builtin_amdgcn_wmma_f32_16x16x32_bf16(
          false, af1[kt], false, bf1, (short)0, acc1, false, false);
      acc2 = __builtin_amdgcn_wmma_f32_16x16x32_bf16(
          false, af2[kt], false, bf2, (short)0, acc2, false, false);
    }
    const float bv1 = bg[col];
    const float bv2 = br[col];
    const size_t obase = (size_t)(tile_m + half * 8) * N + col;
    if (full) {
#pragma unroll
      for (int r = 0; r < 8; ++r)
        out[obase + (size_t)r * N] =
            fmaxf(acc1[r] + bv1, 0.0f) + fmaxf(acc2[r] + bv2, 0.0f);
    } else {
#pragma unroll
      for (int r = 0; r < 8; ++r)
        if (tile_m + half * 8 + r < M)
          out[obase + (size_t)r * N] =
              fmaxf(acc1[r] + bv1, 0.0f) + fmaxf(acc2[r] + bv2, 0.0f);
    }
  }
}

// ---------------------------------------------------------------------------
// Single GEMM (MLP head): out = act(A @ W + bias), act = relu if RELU
// ---------------------------------------------------------------------------
template <int KP, int N, bool RELU>
__global__ __launch_bounds__(256) void wmma_gemm(
    const float* __restrict__ A, const float* __restrict__ W,
    const float* __restrict__ bias, float* __restrict__ out, int M)
{
  constexpr int KT = KP / 32;
  constexpr int NT = N / 16;
  __shared__ __align__(32) bf16 sWt[N * KP];
  for (int i = threadIdx.x; i < N * KP; i += 256) {
    int n = i / KP, k = i % KP;
    sWt[i] = (bf16)W[(size_t)k * N + n];
  }
  __syncthreads();

  const int wave   = threadIdx.x >> 5;
  const int lane   = threadIdx.x & 31;
  const int half   = lane >> 4;
  const int lrow   = lane & 15;
  const int tile_m = blockIdx.x * 128 + wave * 16;
  if (tile_m >= M) return;

  const int  row = tile_m + lrow;
  const bool rv  = row < M;
  const float* ar = A + (size_t)(rv ? row : 0) * KP;

  v16bf afrag[KT];
#pragma unroll
  for (int kt = 0; kt < KT; ++kt) {
    const int b0 = kt * 32 + half * 16;
    v8f xa = *(const v8f*)(ar + b0);
    v8f xb = *(const v8f*)(ar + b0 + 8);
#pragma unroll
    for (int j = 0; j < 8; ++j) {
      afrag[kt][j]     = (bf16)xa[j];
      afrag[kt][j + 8] = (bf16)xb[j];
    }
  }

  const bool full = (tile_m + 16) <= M;

#pragma unroll 1
  for (int nt = 0; nt < NT; ++nt) {
    const int col = nt * 16 + lrow;
    const bf16* bp = sWt + (size_t)col * KP;
    v8f acc = {0.f, 0.f, 0.f, 0.f, 0.f, 0.f, 0.f, 0.f};
#pragma unroll
    for (int kt = 0; kt < KT; ++kt) {
      v16bf bfrag = *(const v16bf*)(bp + kt * 32 + half * 16);
      acc = __builtin_amdgcn_wmma_f32_16x16x32_bf16(
          false, afrag[kt], false, bfrag, (short)0, acc, false, false);
    }
    const float bv = bias[col];
    const size_t obase = (size_t)(tile_m + half * 8) * N + col;
    if (full) {
#pragma unroll
      for (int r = 0; r < 8; ++r) {
        float v = acc[r] + bv;
        if (RELU) v = fmaxf(v, 0.0f);
        out[obase + (size_t)r * N] = v;
      }
    } else {
#pragma unroll
      for (int r = 0; r < 8; ++r) {
        if (tile_m + half * 8 + r < M) {
          float v = acc[r] + bv;
          if (RELU) v = fmaxf(v, 0.0f);
          out[obase + (size_t)r * N] = v;
        }
      }
    }
  }
}

// ---------------------------------------------------------------------------
// Helper kernels
// ---------------------------------------------------------------------------
__global__ void fill_f32(float* __restrict__ p, float v, size_t n) {
  size_t i = (size_t)blockIdx.x * blockDim.x + threadIdx.x;
  if (i < n) p[i] = v;
}

__global__ void pad_rows(const float* __restrict__ in, float* __restrict__ outp,
                         int M, int F, int FP) {
  long long i = (long long)blockIdx.x * blockDim.x + threadIdx.x;
  if (i >= (long long)M * FP) return;
  int r = (int)(i / FP), c = (int)(i % FP);
  outp[i] = (c < F) ? in[(size_t)r * F + c] : 0.0f;
}

__global__ void degree_kernel(const int* __restrict__ src, const int* __restrict__ dst,
                              float* __restrict__ out_deg, float* __restrict__ in_deg, int E) {
  int e = blockIdx.x * blockDim.x + threadIdx.x;
  if (e < E) {
    atomicAdd(&out_deg[src[e]], 1.0f);
    atomicAdd(&in_deg[dst[e]], 1.0f);
  }
}

__global__ void norm_kernel(float* __restrict__ deg, int n) {
  int i = blockIdx.x * blockDim.x + threadIdx.x;
  if (i < n) deg[i] = rsqrtf(fmaxf(deg[i], 1.0f));
}

// agg[dst[e], f] += h[src[e], f] * norm_s[src[e]]; block per edge, thread per feature
__global__ void scatter_edges(const float* __restrict__ h, const float* __restrict__ norm_s,
                              const int* __restrict__ src, const int* __restrict__ dst,
                              float* __restrict__ agg, int F, int hs, int as) {
  int e = blockIdx.x;
  int f = threadIdx.x;
  if (f >= F) return;
  int s = src[e], d = dst[e];
  atomicAdd(&agg[(size_t)d * as + f], h[(size_t)s * hs + f] * norm_s[s]);
}

__global__ void atom_weight(const float* __restrict__ h, const float* __restrict__ w,
                            const float* __restrict__ b, float* __restrict__ aw,
                            int n, int F) {
  int i = blockIdx.x * blockDim.x + threadIdx.x;
  if (i >= n) return;
  float s = b[0];
  for (int f = 0; f < F; ++f) s += h[(size_t)i * F + f] * w[f];
  aw[i] = 1.0f / (1.0f + __expf(-s));
}

// weighted segment-sum + segment-max per graph; block per node, thread per feature
__global__ void readout_kernel(const float* __restrict__ h, const float* __restrict__ aw,
                               const int* __restrict__ gid, float* __restrict__ hsum,
                               float* __restrict__ hmax, int F) {
  int i = blockIdx.x;
  int f = threadIdx.x;
  int g = gid[i];
  float v = h[(size_t)i * F + f];
  atomicAdd(&hsum[(size_t)g * F + f], v * aw[i]);
  float* addr = &hmax[(size_t)g * F + f];
  if (v >= 0.0f) atomicMax((int*)addr, __float_as_int(v));
  else           atomicMin((unsigned int*)addr, __float_as_uint(v));
}

__global__ void concat_kernel(const float* __restrict__ hsum, const float* __restrict__ hmax,
                              float* __restrict__ gfeat, int G, int F) {
  int idx = blockIdx.x * blockDim.x + threadIdx.x;
  if (idx >= G * 2 * F) return;
  int g = idx / (2 * F), f = idx % (2 * F);
  gfeat[idx] = (f < F) ? hsum[(size_t)g * F + f] : hmax[(size_t)g * F + (f - F)];
}

// insert zero rows at idx_wo (sorted ascending); writes every output element
__global__ void expand_kernel(const float* __restrict__ latent, const int* __restrict__ idx_wo,
                              float* __restrict__ out, int Kz, int D) {
  int t = blockIdx.x;
  int f = threadIdx.x;
  int cnt = 0; bool masked = false;
  for (int j = 0; j < Kz; ++j) {
    int z = idx_wo[j];
    if (z == t) masked = true;
    if (z < t)  cnt++;
  }
  out[(size_t)t * D + f] = masked ? 0.0f : latent[(size_t)(t - cnt) * D + f];
}

// ---------------------------------------------------------------------------
// Launch
// ---------------------------------------------------------------------------
extern "C" void kernel_launch(void* const* d_in, const int* in_sizes, int n_in,
                              void* d_out, int out_size, void* d_ws, size_t ws_size,
                              hipStream_t stream) {
  const float* node_feats = (const float*)d_in[0];
  // d_in[1] edge_feats: unused by the GCN
  const float* W1  = (const float*)d_in[2];
  const float* b1  = (const float*)d_in[3];
  const float* Wr1 = (const float*)d_in[4];
  const float* br1 = (const float*)d_in[5];
  const float* W2  = (const float*)d_in[6];
  const float* b2  = (const float*)d_in[7];
  const float* Wr2 = (const float*)d_in[8];
  const float* br2 = (const float*)d_in[9];
  const float* w_atom = (const float*)d_in[10];
  const float* b_atom = (const float*)d_in[11];
  const float* Wp1 = (const float*)d_in[12];
  const float* bp1 = (const float*)d_in[13];
  const float* Wp2 = (const float*)d_in[14];
  const float* bp2 = (const float*)d_in[15];
  const int* src = (const int*)d_in[16];
  const int* dst = (const int*)d_in[17];
  const int* gid = (const int*)d_in[18];
  const int* idx_wo = (const int*)d_in[19];

  const int N   = in_sizes[18];          // 100000 nodes
  const int E   = in_sizes[16];          // 400000 edges
  const int Kz  = in_sizes[19];          // 3 zero rows
  const int NF  = in_sizes[0] / N;       // 74
  const int NFP = 96;                    // 74 padded to multiple of 32
  const int HID = in_sizes[3];           // 128
  const int DIM = in_sizes[15];          // 256
  const int G   = out_size / DIM - Kz;   // 2500

  // workspace layout (fp32); all offsets multiples of 4 floats (16B aligned)
  float* w = (float*)d_ws;
  size_t off = 0;
  float* norm_s   = w + off; off += N;
  float* norm_d   = w + off; off += N;
  float* node_pad = w + off; off += (size_t)N * NFP;  // node_feats padded 74->96
  float* agg      = w + off; off += (size_t)N * HID;  // stride 96 (L1) / 128 (L2)
  float* h1       = w + off; off += (size_t)N * HID;
  float* h2       = w + off; off += (size_t)N * HID;
  float* aw       = w + off; off += N;
  float* hsum     = w + off; off += (size_t)G * HID;
  float* hmax     = w + off; off += (size_t)G * HID;
  float* gfeat    = w + off; off += (size_t)G * 2 * HID;
  float* tmp      = w + off; off += (size_t)G * HID;
  float* latent   = w + off; off += (size_t)G * DIM;
  (void)ws_size; (void)n_in;

  auto cdiv = [](long long a, long long b) { return (unsigned)((a + b - 1) / b); };

  // degrees -> norms (norm_s, norm_d contiguous)
  fill_f32<<<cdiv(2LL * N, 256), 256, 0, stream>>>(norm_s, 0.0f, (size_t)2 * N);
  degree_kernel<<<cdiv(E, 256), 256, 0, stream>>>(src, dst, norm_s, norm_d, E);
  norm_kernel<<<cdiv(2LL * N, 256), 256, 0, stream>>>(norm_s, 2 * N);

  // padded copy of node features (zeros in cols 74..95)
  pad_rows<<<cdiv((long long)N * NFP, 256), 256, 0, stream>>>(node_feats, node_pad, N, NF, NFP);

  // ---- GCN layer 1 (74 -> 128), A-stride 96, fused gc+residual ----
  fill_f32<<<cdiv((long long)N * NFP, 256), 256, 0, stream>>>(agg, 0.0f, (size_t)N * NFP);
  scatter_edges<<<E, 128, 0, stream>>>(node_pad, norm_s, src, dst, agg, NF, NFP, NFP);
  wmma_dual_gemm<96, 128><<<cdiv(N, 128), 256, 0, stream>>>(
      agg, norm_d, W1, b1, node_pad, Wr1, br1, h1, N, NF);

  // ---- GCN layer 2 (128 -> 128), fused gc+residual ----
  fill_f32<<<cdiv((long long)N * HID, 256), 256, 0, stream>>>(agg, 0.0f, (size_t)N * HID);
  scatter_edges<<<E, 128, 0, stream>>>(h1, norm_s, src, dst, agg, HID, HID, HID);
  wmma_dual_gemm<128, 128><<<cdiv(N, 128), 256, 0, stream>>>(
      agg, norm_d, W2, b2, h1, Wr2, br2, h2, N, HID);

  // ---- WeightedSumAndMax readout ----
  atom_weight<<<cdiv(N, 256), 256, 0, stream>>>(h2, w_atom, b_atom, aw, N, HID);
  fill_f32<<<cdiv((long long)G * HID, 256), 256, 0, stream>>>(hsum, 0.0f, (size_t)G * HID);
  fill_f32<<<cdiv((long long)G * HID, 256), 256, 0, stream>>>(hmax, -INFINITY, (size_t)G * HID);
  readout_kernel<<<N, HID, 0, stream>>>(h2, aw, gid, hsum, hmax, HID);
  concat_kernel<<<cdiv((long long)G * 2 * HID, 256), 256, 0, stream>>>(hsum, hmax, gfeat, G, HID);

  // ---- MLP head ----
  wmma_gemm<256, 128, true><<<cdiv(G, 128), 256, 0, stream>>>(gfeat, Wp1, bp1, tmp, G);
  wmma_gemm<128, 256, false><<<cdiv(G, 128), 256, 0, stream>>>(tmp, Wp2, bp2, latent, G);

  // ---- insert zero rows ----
  expand_kernel<<<G + Kz, DIM, 0, stream>>>(latent, idx_wo, (float*)d_out, Kz, DIM);
}
